// Attention_52012053955159
// MI455X (gfx1250) — matrix operand order repaired and checked by reading, not compiled, not run on hardware
//
#include <hip/hip_runtime.h>

// ---------------------------------------------------------------------------
// Masked attention without softmax, algebraically collapsed:
//   values = where(m, -1e9, Q K^T) V
//          = mask_i ? -1e9 * sum_j V[j]
//                   : Q_i @ (K_sel^T V)  - 1e9 * sum_{masked j} V[j]
// Heavy math on v_wmma_f32_16x16x32_bf16 (fp32 accumulate); GEMM tiles are
// staged global->LDS by the Tensor Data Mover (tensor_load_to_lds) with
// descriptor-side padding, double buffered against the WMMA loop.
// Workspace (~93 MB): xb | WqkvT | WoT | QKV | A^T | Vm | Vall | Vals
// ---------------------------------------------------------------------------

typedef __bf16 bf16_t;
typedef __attribute__((ext_vector_type(16))) __bf16 v16bf;
typedef __attribute__((ext_vector_type(8)))  __bf16 v8bf;
typedef __attribute__((ext_vector_type(8)))  float  v8f;
typedef __attribute__((ext_vector_type(4)))  unsigned int u32x4;
typedef __attribute__((ext_vector_type(8)))  int i32x8;
typedef __attribute__((ext_vector_type(4)))  int i32x4;

#define BB   4
#define SS   2048
#define DIN  1024
#define HH   16
#define DK   64
#define DV   64
#define DOUT 1024
#define NQKV 3072           // packed per row: n = h*192 + {q:0..63, k:64..127, v:128..191}
#define NEGC (-1e9f)
#define LPAD 48             // LDS row stride (elements): 96B = 64B row + 32B pad

static __device__ __forceinline__ v8f wmma_bf16(v16bf a, v16bf b, v8f c) {
  return __builtin_amdgcn_wmma_f32_16x16x32_bf16(false, a, false, b, (short)0, c,
                                                 false, false);
}

// A-operand fragment: lane holds row m = lane&15; K elements
// {0..7,16..23} for lanes 0-15, {8..15,24..31} for lanes 16-31 (ISA 7.12.2).
static __device__ __forceinline__ v16bf frag_a(const bf16_t* p0, int stride, int lane) {
  const int m    = lane & 15;
  const int koff = (lane & 16) ? 8 : 0;
  const bf16_t* p = p0 + m * stride + koff;
  union { v16bf v; v8bf h[2]; } u;
  u.h[0] = *(const v8bf*)(p);
  u.h[1] = *(const v8bf*)(p + 16);
  return u.v;
}

// B-operand fragment: lane holds col n = lane&15; K elements
// {0..7,16..23} for lanes 0-15, {8..15,24..31} for lanes 16-31. Memory must be [N][K] (transposed B).
static __device__ __forceinline__ v16bf frag_b(const bf16_t* p0, int stride, int lane) {
  const int n    = lane & 15;
  const int koff = (lane & 16) ? 8 : 0;
  const bf16_t* p = p0 + n * stride + koff;
  union { v16bf v; v8bf h[2]; } u;
  u.h[0] = *(const v8bf*)(p);
  u.h[1] = *(const v8bf*)(p + 16);
  return u.v;
}

// ---------------- Tensor Data Mover: 2-D bf16 tile -> padded LDS -----------
// D# per CDNA5 ISA. Tile: tile1 rows x 32 cols of 2-byte
// elements (64B rows), source row stride = strideElems. LDS padding: after
// every 16 DWORDs (64B = one row) insert 8 DWORDs (32B) -> row stride 96B,
// matching the LPAD=48 layout consumed by frag_a/frag_b.
static __device__ __forceinline__ void tdm_load_tile(
    unsigned lds_off, const bf16_t* src, unsigned tile1,
    unsigned long long strideElems) {
  const unsigned long long ga = (unsigned long long)(size_t)(const void*)src;
  u32x4 g0;
  g0[0] = 1u;                                        // count=1, no gather
  g0[1] = lds_off;                                   // lds_addr (bytes)
  g0[2] = (unsigned)ga;                              // global_addr[31:0]
  g0[3] = (unsigned)((ga >> 32) & 0x1FFFFFFu)        // global_addr[56:32]
          | (2u << 30);                              // type = 2 (image)
  i32x8 g1;
  g1[0] = (int)((1u << 16)                           // data_size = 2 bytes
                | (1u << 20)                         // pad_enable
                | (3u << 22)                         // pad_interval: 16 DWORDs
                | (7u << 25));                       // pad_amount:   8 DWORDs
  g1[1] = (int)(32u << 16);                          // tensor_dim0 = 32
  g1[2] = (int)((tile1 & 0xFFFFu) << 16);            // tensor_dim1 = tile1
  g1[3] = (int)(32u << 16);                          // tile_dim0 = 32
  g1[4] = (int)(tile1 & 0xFFFFu);                    // tile_dim1, tile_dim2=0
  g1[5] = (int)(unsigned)strideElems;                // dim0_stride[31:0]
  g1[6] = (int)((strideElems >> 32) & 0xFFFFu);      // dim0_stride[47:32]
  g1[7] = 0;
  const i32x4 gz = {0, 0, 0, 0};
#if defined(__clang_major__) && __clang_major__ >= 23
  const i32x8 gz8 = {0, 0, 0, 0, 0, 0, 0, 0};
  __builtin_amdgcn_tensor_load_to_lds(g0, g1, gz, gz, gz8, 0);
#else
  __builtin_amdgcn_tensor_load_to_lds(g0, g1, gz, gz, 0);
#endif
}

// ------------------------------- converters --------------------------------

__global__ void cvt_f32_bf16(const float* __restrict__ in, bf16_t* __restrict__ out,
                             int n) {
  int i = blockIdx.x * blockDim.x + threadIdx.x;
  if (i < n) out[i] = (bf16_t)in[i];
}

// WqkvT[n][d], n = h*192 + j : q(0..63), k(64..127), v(128..191)
__global__ void build_wqkvT(const float* __restrict__ Wq, const float* __restrict__ Wk,
                            const float* __restrict__ Wv, bf16_t* __restrict__ out) {
  int i = blockIdx.x * blockDim.x + threadIdx.x;   // over NQKV*DIN
  int d = i & (DIN - 1);
  int n = i >> 10;
  int h = n / 192;
  int j = n - h * 192;
  float v;
  if (j < 64)        v = Wq[((long)h * DIN + d) * DK + j];
  else if (j < 128)  v = Wk[((long)h * DIN + d) * DK + (j - 64)];
  else               v = Wv[((long)h * DIN + d) * DV + (j - 128)];
  out[i] = (bf16_t)v;
}

// WoT[d][h*64+v] = W_o[h][v][d]
__global__ void build_woT(const float* __restrict__ Wo, bf16_t* __restrict__ out) {
  int i = blockIdx.x * blockDim.x + threadIdx.x;   // over DOUT*(HH*DV)
  int c = i & 1023;
  int d = i >> 10;
  out[i] = (bf16_t)Wo[(long)c * DOUT + d];         // c = h*64+v -> W_o[h][v][d]
}

// --------------------------- tiled WMMA GEMM -------------------------------
// C[M x N] = A[M x K] * B[K x N], A row-major, B given transposed BT[N][K].
// Block: 256 threads = 8 wave32, tile 128(M) x 256(N); each wave 64x64 = 4x4
// v_wmma_f32_16x16x32_bf16 accumulators. A/BT K-slices DMA'd into padded LDS
// by the TDM, double buffered (wave 0 issues, s_wait_tensorcnt, barrier).
template <bool OUT_BF16>
__global__ __launch_bounds__(256) void gemm_tile(
    const bf16_t* __restrict__ A, const bf16_t* __restrict__ BT,
    void* __restrict__ Cv, int lda, int ldbt, int ldc,
    long aBatch, long cBatch, int Kd) {
  __shared__ __align__(128) bf16_t lsA[2][128 * LPAD];
  __shared__ __align__(128) bf16_t lsB[2][256 * LPAD];

  const int tid  = threadIdx.x;
  const int lane = tid & 31;
  const int wave = tid >> 5;
  const int wm   = (wave & 1) * 64;      // 2 waves along M
  const int wn   = (wave >> 1) * 64;     // 4 waves along N
  const int n0   = blockIdx.x * 256;
  const int m0   = blockIdx.y * 128;
  const long ab  = (long)blockIdx.z * aBatch;
  const long cb  = (long)blockIdx.z * cBatch;

  const unsigned aOff0 = (unsigned)(size_t)(void*)&lsA[0][0];
  const unsigned aOff1 = (unsigned)(size_t)(void*)&lsA[1][0];
  const unsigned bOff0 = (unsigned)(size_t)(void*)&lsB[0][0];
  const unsigned bOff1 = (unsigned)(size_t)(void*)&lsB[1][0];
  const bf16_t* Abase = A + ab + (long)m0 * lda;
  const bf16_t* Bbase = BT + (long)n0 * ldbt;

  const v8f vzero = {0.f, 0.f, 0.f, 0.f, 0.f, 0.f, 0.f, 0.f};
  v8f acc[4][4];
#pragma unroll
  for (int i = 0; i < 4; ++i)
#pragma unroll
    for (int j = 0; j < 4; ++j) acc[i][j] = vzero;

  const int nT = Kd / 32;
  if (wave == 0) {                        // prologue: DMA slice 0 -> buffer 0
    tdm_load_tile(aOff0, Abase, 128, (unsigned long long)lda);
    tdm_load_tile(bOff0, Bbase, 256, (unsigned long long)ldbt);
  }

  for (int t = 0; t < nT; ++t) {
    const int cur = t & 1;
    if (wave == 0) {
      if (t + 1 < nT) {                   // DMA next slice into other buffer
        const int k1 = (t + 1) * 32;
        tdm_load_tile(cur ? aOff0 : aOff1, Abase + k1, 128,
                      (unsigned long long)lda);
        tdm_load_tile(cur ? bOff0 : bOff1, Bbase + k1, 256,
                      (unsigned long long)ldbt);
        __builtin_amdgcn_s_wait_tensorcnt(2);   // current slice's 2 ops done
      } else {
        __builtin_amdgcn_s_wait_tensorcnt(0);
      }
    }
    __syncthreads();

    v16bf af[4], bq[4];
#pragma unroll
    for (int mi = 0; mi < 4; ++mi)
      af[mi] = frag_a(&lsA[cur][(wm + mi * 16) * LPAD], LPAD, lane);
#pragma unroll
    for (int ni = 0; ni < 4; ++ni)
      bq[ni] = frag_b(&lsB[cur][(wn + ni * 16) * LPAD], LPAD, lane);
#pragma unroll
    for (int mi = 0; mi < 4; ++mi)
#pragma unroll
      for (int ni = 0; ni < 4; ++ni)
        acc[mi][ni] = wmma_bf16(af[mi], bq[ni], acc[mi][ni]);
    __syncthreads();                      // buffer reusable for t+2's DMA
  }

  // C fragment layout: VGPR r -> M = r + (lane<16 ? 0 : 8), N = lane&15
  const int rsel = (lane & 16) ? 8 : 0;
  const int cn   = lane & 15;
  for (int mi = 0; mi < 4; ++mi)
    for (int ni = 0; ni < 4; ++ni)
#pragma unroll
      for (int r = 0; r < 8; ++r) {
        long row = m0 + wm + mi * 16 + r + rsel;
        long col = n0 + wn + ni * 16 + cn;
        float v  = acc[mi][ni][r];
        if (OUT_BF16) ((bf16_t*)Cv)[cb + row * ldc + col] = (bf16_t)v;
        else          ((float*)Cv)[cb + row * ldc + col]  = v;
      }
}

// --------------------- per-(b,h): A = K_sel^T V + V sums -------------------
__global__ __launch_bounds__(32) void attn_kv_reduce(
    const bf16_t* __restrict__ QKVc, const unsigned char* __restrict__ mask,
    bf16_t* __restrict__ ATm, float* __restrict__ Vm, float* __restrict__ Vall) {
  __shared__ bf16_t lK[64 * LPAD];   // [k][j] transposed, mask applied to K
  __shared__ bf16_t lV[64 * LPAD];   // [v][j] transposed, raw
  __shared__ float  mskS[32];

  const int lane = threadIdx.x;
  const int bh   = blockIdx.x;
  const int b    = bh >> 4;
  const bf16_t* Kp = QKVc + (long)b * SS * NQKV + (bh & 15) * 192 + DK;
  const bf16_t* Vp = Kp + DK;

  const v8f vzero = {0.f, 0.f, 0.f, 0.f, 0.f, 0.f, 0.f, 0.f};
  v8f acc[4][4];
#pragma unroll
  for (int i = 0; i < 4; ++i)
#pragma unroll
    for (int j = 0; j < 4; ++j) acc[i][j] = vzero;
  float sA0 = 0.f, sA1 = 0.f, sM0 = 0.f, sM1 = 0.f;

  for (int j0 = 0; j0 < SS; j0 += 32) {
    const int   j   = j0 + lane;
    const float m   = mask[b * SS + j] ? 1.f : 0.f;
    const float sel = 1.f - m;
    mskS[lane] = m;
    const bf16_t* kr = Kp + (long)j * NQKV;
    const bf16_t* vr = Vp + (long)j * NQKV;
#pragma unroll
    for (int k = 0; k < 64; ++k) {        // transpose rows into LDS columns
      lK[k * LPAD + lane] = (bf16_t)((float)kr[k] * sel);
      lV[k * LPAD + lane] = vr[k];
    }
    __syncthreads();

    for (int jj = 0; jj < 32; ++jj) {     // V column sums (all j / masked j)
      const float mm = mskS[jj];
      const float v0 = (float)lV[lane * LPAD + jj];
      const float v1 = (float)lV[(lane + 32) * LPAD + jj];
      sA0 += v0; sA1 += v1;
      sM0 += mm * v0; sM1 += mm * v1;
    }

    v16bf ak[4], bv[4];
#pragma unroll
    for (int mi = 0; mi < 4; ++mi) ak[mi] = frag_a(&lK[mi * 16 * LPAD], LPAD, lane);
#pragma unroll
    for (int ni = 0; ni < 4; ++ni) bv[ni] = frag_b(&lV[ni * 16 * LPAD], LPAD, lane);
#pragma unroll
    for (int mi = 0; mi < 4; ++mi)
#pragma unroll
      for (int ni = 0; ni < 4; ++ni)
        acc[mi][ni] = wmma_bf16(ak[mi], bv[ni], acc[mi][ni]);
    __syncthreads();
  }

  // store A transposed: ATm[bh][v][k] (B-operand layout for the next GEMM)
  const int rsel = (lane & 16) ? 8 : 0;
  const int cn   = lane & 15;
  bf16_t* Ab = ATm + (long)bh * (DK * DV);
  for (int mi = 0; mi < 4; ++mi)
    for (int ni = 0; ni < 4; ++ni)
#pragma unroll
      for (int r = 0; r < 8; ++r) {
        int kk = mi * 16 + r + rsel;
        int vv = ni * 16 + cn;
        Ab[vv * DK + kk] = (bf16_t)acc[mi][ni][r];
      }
  Vall[bh * 64 + lane]      = sA0;
  Vall[bh * 64 + lane + 32] = sA1;
  Vm[bh * 64 + lane]        = sM0;
  Vm[bh * 64 + lane + 32]   = sM1;
}

// ------------------- values = Q @ A  (+ mask fixups), bf16 -----------------
__global__ __launch_bounds__(128) void attn_apply(
    const bf16_t* __restrict__ QKVc, const bf16_t* __restrict__ ATm,
    const float* __restrict__ Vm, const float* __restrict__ Vall,
    const unsigned char* __restrict__ mask, bf16_t* __restrict__ Vals) {
  const int tid = threadIdx.x, lane = tid & 31, wave = tid >> 5;
  const int bh = blockIdx.y, b = bh >> 4, h = bh & 15;
  const int m0 = blockIdx.x * 256 + wave * 64;
  const bf16_t* Qp = QKVc + (long)b * SS * NQKV + h * 192;  // Q[s][k], stride NQKV
  const bf16_t* Ap = ATm + (long)bh * (DK * DV);            // [v][k], stride DK

  const v8f vzero = {0.f, 0.f, 0.f, 0.f, 0.f, 0.f, 0.f, 0.f};
  v8f acc[4][4];
#pragma unroll
  for (int i = 0; i < 4; ++i)
#pragma unroll
    for (int j = 0; j < 4; ++j) acc[i][j] = vzero;

#pragma unroll
  for (int k0 = 0; k0 < DK; k0 += 32) {
    v16bf aq[4], bq[4];
#pragma unroll
    for (int mi = 0; mi < 4; ++mi)
      aq[mi] = frag_a(Qp + (long)(m0 + mi * 16) * NQKV + k0, NQKV, lane);
#pragma unroll
    for (int ni = 0; ni < 4; ++ni)
      bq[ni] = frag_b(Ap + (ni * 16) * DK + k0, DK, lane);
#pragma unroll
    for (int mi = 0; mi < 4; ++mi)
#pragma unroll
      for (int ni = 0; ni < 4; ++ni)
        acc[mi][ni] = wmma_bf16(aq[mi], bq[ni], acc[mi][ni]);
  }

  const int rsel = (lane & 16) ? 8 : 0;
  const int cn   = lane & 15;
  for (int ni = 0; ni < 4; ++ni) {
    const int   vv = ni * 16 + cn;
    const float vm = Vm[bh * 64 + vv] * NEGC;    // -1e9 * sum_{masked j} V
    const float va = Vall[bh * 64 + vv] * NEGC;  // -1e9 * sum_j V
    for (int mi = 0; mi < 4; ++mi)
#pragma unroll
      for (int r = 0; r < 8; ++r) {
        const int  s  = m0 + mi * 16 + r + rsel;
        const bool mk = mask[b * SS + s] != 0;
        const float val = mk ? va : (acc[mi][ni][r] + vm);
        Vals[((long)b * SS + s) * (HH * DV) + h * DV + vv] = (bf16_t)val;
      }
  }
}

// ------------------------------- launcher ----------------------------------
extern "C" void kernel_launch(void* const* d_in, const int* in_sizes, int n_in,
                              void* d_out, int out_size, void* d_ws, size_t ws_size,
                              hipStream_t stream) {
  (void)in_sizes; (void)n_in; (void)out_size; (void)ws_size;
  const float* x  = (const float*)d_in[0];
  const unsigned char* mask = (const unsigned char*)d_in[1];  // jax bool = 1 byte
  const float* Wq = (const float*)d_in[2];
  const float* Wk = (const float*)d_in[3];
  const float* Wv = (const float*)d_in[4];
  const float* Wo = (const float*)d_in[5];
  float* out = (float*)d_out;

  char*  ws = (char*)d_ws;
  size_t o  = 0;
  bf16_t* xb    = (bf16_t*)(ws + o); o += (size_t)BB * SS * DIN * 2;        // 16 MB
  bf16_t* WqkvT = (bf16_t*)(ws + o); o += (size_t)NQKV * DIN * 2;           //  6 MB
  bf16_t* WoT   = (bf16_t*)(ws + o); o += (size_t)DOUT * HH * DV * 2;       //  2 MB
  bf16_t* QKVc  = (bf16_t*)(ws + o); o += (size_t)BB * SS * NQKV * 2;       // 48 MB
  bf16_t* ATm   = (bf16_t*)(ws + o); o += (size_t)BB * HH * DK * DV * 2;    // .5 MB
  float*  Vm    = (float*)(ws + o);  o += (size_t)BB * HH * DV * 4;
  float*  Vall  = (float*)(ws + o);  o += (size_t)BB * HH * DV * 4;
  bf16_t* Vals  = (bf16_t*)(ws + o); o += (size_t)BB * SS * HH * DV * 2;    // 16 MB

  // 1) fp32 -> bf16 conversions / weight packing+transposition
  cvt_f32_bf16<<<(BB * SS * DIN) / 256, 256, 0, stream>>>(x, xb, BB * SS * DIN);
  build_wqkvT<<<(NQKV * DIN) / 256, 256, 0, stream>>>(Wq, Wk, Wv, WqkvT);
  build_woT<<<(DOUT * HH * DV) / 256, 256, 0, stream>>>(Wo, WoT);

  // 2) fused QKV projection: [S x DIN] @ [DIN x 3072] per batch (TDM-staged)
  gemm_tile<true><<<dim3(NQKV / 256, SS / 128, BB), 256, 0, stream>>>(
      xb, WqkvT, QKVc, DIN, DIN, NQKV, (long)SS * DIN, (long)SS * NQKV, DIN);

  // 3) per-(b,h) 64x64 A = K_sel^T V, plus V column sums
  attn_kv_reduce<<<BB * HH, 32, 0, stream>>>(QKVc, mask, ATm, Vm, Vall);

  // 4) values = Q @ A with mask fixups, packed [b][s][h*64+v] bf16
  attn_apply<<<dim3(SS / 256, BB * HH), 128, 0, stream>>>(QKVc, ATm, Vm, Vall,
                                                          mask, Vals);

  // 5) output projection: [S x 1024] @ [1024 x 1024] per batch, fp32 out
  gemm_tile<false><<<dim3(DOUT / 256, SS / 128, BB), 256, 0, stream>>>(
      Vals, WoT, out, HH * DV, HH * DV, DOUT, (long)SS * HH * DV,
      (long)SS * DOUT, HH * DV);
}
